// QueryFocusedFullyAttentionNetwork_34969623724739
// MI455X (gfx1250) — matrix operand order repaired and checked by reading, not compiled, not run on hardware
//
#include <hip/hip_runtime.h>

// ---------------------------------------------------------------------------
// QueryFocusedFullyAttentionNetwork for MI455X (gfx1250, wave32, WMMA + TDM).
// All large GEMMs use v_wmma_f32_16x16x32_f16 (f16 in, f32 accum) with the
// B panel staged into LDS by the Tensor Data Mover (tensor_load_to_lds,
// TENSORcnt).  Requires ws_size >= ~480 MB (bump-allocated below).
// ---------------------------------------------------------------------------

typedef _Float16 half_t;
typedef __attribute__((ext_vector_type(16))) _Float16 v16h;
typedef __attribute__((ext_vector_type(8)))  _Float16 v8h;
typedef __attribute__((ext_vector_type(8)))  float    v8f;
typedef __attribute__((ext_vector_type(4)))  unsigned int v4u;
typedef __attribute__((ext_vector_type(8)))  int v8i;
typedef __attribute__((ext_vector_type(4)))  int v4i;

#define BDIM 8
#define SDIM 32
#define FDIM 64
#define CDIM 1000
#define CP   1024
#define EDIM 256
#define DKDIM 1024
#define HDIM 8
#define HDD  125
#define DFFD 2048
#define LNUM 6
#define NROW (BDIM * SDIM * FDIM)   /* 16384 token rows */
#define EPSV 1e-5f
#define KCHUNK 128                  /* K rows of B staged per TDM transfer */

// ---------------------------------------------------------------------------
// frame_fea [B,F,S,C] f32 -> x [n=b*S+s, f, CP] (f32 residual copy + f16 copy)
// ---------------------------------------------------------------------------
__global__ void k_prep(const float* __restrict__ src,
                       float* __restrict__ x32, half_t* __restrict__ xh)
{
    size_t tid = (size_t)blockIdx.x * blockDim.x + threadIdx.x;
    size_t total = (size_t)NROW * CP;
    if (tid >= total) return;
    int c   = (int)(tid % CP);
    int row = (int)(tid / CP);
    int f = row % FDIM;
    int n = row / FDIM;
    int s = n % SDIM;
    int b = n / SDIM;
    float v = 0.f;
    if (c < CDIM)
        v = src[(((size_t)b * FDIM + f) * SDIM + s) * CDIM + c];
    x32[tid] = v;
    xh[tid]  = (half_t)v;
}

// Generic f32 -> f16 pack with zero padding.
__global__ void k_pack_h(const float* __restrict__ src, int lds_,
                         half_t* __restrict__ dst, int ldd,
                         int Rs, int Cs, int Rd, int Cd)
{
    size_t tid = (size_t)blockIdx.x * blockDim.x + threadIdx.x;
    size_t total = (size_t)Rd * Cd;
    if (tid >= total) return;
    int c = (int)(tid % Cd);
    int r = (int)(tid / Cd);
    float v = (r < Rs && c < Cs) ? src[(size_t)r * lds_ + c] : 0.f;
    dst[(size_t)r * ldd + c] = (half_t)v;
}

__global__ void k_pad_f32(const float* __restrict__ src, float* __restrict__ dst,
                          int n, int np)
{
    int i = blockIdx.x * 256 + threadIdx.x;
    if (i < np) dst[i] = (i < n) ? src[i] : 0.f;
}

// ---------------------------------------------------------------------------
// WMMA GEMM with TDM-staged B panel.
// D[M,N] = act(A[M,K] * B[K,N] + bias[N]); f16 A/B, f32 accumulate.
// Block = 256 threads = 8 waves; each wave computes a 32x16 tile (2 accums
// sharing one B fragment); block covers 256 M-rows x 16 N-cols.
// grid = (N/16, M/256).  K % 128 == 0, N % 16 == 0, M % 256 == 0.
//
// Per K-chunk (128 rows): wave 0 issues tensor_load_to_lds with a 2D D#
// (tile 16 x 128, 2-byte elements, row stride N), waits TENSORcnt==0,
// then the block barriers and computes from LDS.
//
// A-fragment (ISA 16-bit A 16x32): lanes 0-15 hold M=lane with K chunks
// [k..k+7]+[k+16..k+23]; lanes 16-31 hold the +8 chunks.
// B-fragment: lane&15 -> N column, 16 vector elements -> K rows (same K order as A).
// C/D: VGPR r, lane l -> M = r + 8*(l>=16), N = l&15.
// ---------------------------------------------------------------------------
__global__ __launch_bounds__(256) void k_gemm(
    const half_t* __restrict__ A, const half_t* __restrict__ B,
    const float* __restrict__ bias, float* __restrict__ Cf,
    half_t* __restrict__ Ch, int M, int N, int K, int act)
{
    __shared__ __align__(32) half_t Bs[KCHUNK * 16];

    const int lane  = threadIdx.x & 31;
    const int wave  = threadIdx.x >> 5;
    const int ntile = blockIdx.x * 16;
    const int mbase = blockIdx.y * 256 + wave * 32;
    const int mrow  = lane & 15;
    const int hi    = lane >> 4;

    const half_t* Arow0 = A + (size_t)(mbase + mrow) * K;
    const half_t* Arow1 = A + (size_t)(mbase + 16 + mrow) * K;

    v8f acc0 = {};
    v8f acc1 = {};

    const unsigned int lds_off = (unsigned int)(uintptr_t)(&Bs[0]);

    for (int k0 = 0; k0 < K; k0 += KCHUNK) {
        if (threadIdx.x < 32) {
            // ---- Tensor DMA descriptor (CDNA5 ISA) ----
            unsigned long long ga =
                (unsigned long long)(uintptr_t)(B + (size_t)k0 * N + ntile);
            v4u g0;
            g0.x = 1u;                                   // count=1, user mode
            g0.y = lds_off;                              // lds_addr
            g0.z = (unsigned int)ga;                     // global_addr[31:0]
            g0.w = (unsigned int)(ga >> 32) | 0x80000000u; // ga[56:32] | type=2
            v8i g1;
            g1[0] = 0x10000;                 // wg_mask=0, data_size=1 (2 bytes)
            g1[1] = (int)(16u << 16);        // tensor_dim0 = 16 (low16)
            g1[2] = (int)((unsigned)KCHUNK << 16); // dim0 hi=0 | tensor_dim1 lo16
            g1[3] = (int)(16u << 16);        // dim1 hi=0 | tile_dim0 = 16
            g1[4] = KCHUNK;                  // tile_dim1 = 128, tile_dim2 = 0
            g1[5] = N;                       // tensor_dim0_stride low32 (elems)
            g1[6] = 0;                       // stride hi | dim1_stride lo
            g1[7] = 0;
            v4i gz = {0, 0, 0, 0};
            asm volatile("tensor_load_to_lds %0, %1, %2, %3"
                         :: "s"(g0), "s"(g1), "s"(gz), "s"(gz)
                         : "memory");
            __builtin_amdgcn_s_wait_tensorcnt(0);
        }
        __syncthreads();

#pragma unroll
        for (int kk = 0; kk < KCHUNK; kk += 32) {
            const int kg = k0 + kk;
            __builtin_prefetch(Arow0 + kg + 256, 0, 1);   // global_prefetch_b8
            __builtin_prefetch(Arow1 + kg + 256, 0, 1);
            v8h a0lo = *(const v8h*)(Arow0 + kg + hi * 8);
            v8h a0hi = *(const v8h*)(Arow0 + kg + 16 + hi * 8);
            v8h a1lo = *(const v8h*)(Arow1 + kg + hi * 8);
            v8h a1hi = *(const v8h*)(Arow1 + kg + 16 + hi * 8);
            v16h a0, a1;
#pragma unroll
            for (int i = 0; i < 8; ++i) {
                a0[i] = a0lo[i]; a0[i + 8] = a0hi[i];
                a1[i] = a1lo[i]; a1[i + 8] = a1hi[i];
            }
            v16h b;
#pragma unroll
            for (int i = 0; i < 8; ++i) {
                b[i]     = Bs[(kk + hi * 8 + i) * 16 + mrow];
                b[i + 8] = Bs[(kk + 16 + hi * 8 + i) * 16 + mrow];
            }
            acc0 = __builtin_amdgcn_wmma_f32_16x16x32_f16(
                false, a0, false, b, (short)0, acc0, false, false);
            acc1 = __builtin_amdgcn_wmma_f32_16x16x32_f16(
                false, a1, false, b, (short)0, acc1, false, false);
        }
        __syncthreads();
    }

    const int n  = ntile + mrow;
    const float bn = bias ? bias[n] : 0.f;
#pragma unroll
    for (int r = 0; r < 8; ++r) {
        int m0 = mbase + r + hi * 8;
        int m1 = m0 + 16;
        float v0 = acc0[r] + bn;
        float v1 = acc1[r] + bn;
        if (act) { v0 = fmaxf(v0, 0.f); v1 = fmaxf(v1, 0.f); }
        if (Cf) { Cf[(size_t)m0 * N + n] = v0; Cf[(size_t)m1 * N + n] = v1; }
        if (Ch) { Ch[(size_t)m0 * N + n] = (half_t)v0; Ch[(size_t)m1 * N + n] = (half_t)v1; }
    }
}

// ---------------------------------------------------------------------------
// Encoder self-attention: block per (n, h, qrow); 64 threads = 64 keys.
// QKVh row layout: [q(0..1023) | k(1024..2047) | v(2048..3071)], head h at
// offset h*125 inside each 1024-wide section.  Output heads concatenated to
// Oh [NROW, CP] (cols 1000..1023 zeroed by the h==0 blocks).
// ---------------------------------------------------------------------------
__global__ __launch_bounds__(64) void k_attn(const half_t* __restrict__ qkv,
                                             half_t* __restrict__ oh)
{
    const float scale = 0.0894427190999916f;  // 1/sqrt(125)
    int blk  = blockIdx.x;
    int qrow = blk % FDIM; blk /= FDIM;
    int h    = blk % HDIM;
    int n    = blk / HDIM;
    int t    = threadIdx.x;                   // key index

    const half_t* base = qkv + (size_t)n * FDIM * 3072;
    const half_t* qp = base + (size_t)qrow * 3072 + h * HDD;
    const half_t* kp = base + (size_t)t * 3072 + 1024 + h * HDD;
    float dot = 0.f;
    for (int d = 0; d < HDD; ++d) dot += (float)qp[d] * (float)kp[d];
    float sval = dot * scale;

    __shared__ float sc[64];
    __shared__ float red[64];
    red[t] = sval; __syncthreads();
    for (int o = 32; o > 0; o >>= 1) { if (t < o) red[t] = fmaxf(red[t], red[t + o]); __syncthreads(); }
    float mx = red[0]; __syncthreads();
    float e = expf(sval - mx);
    red[t] = e; __syncthreads();
    for (int o = 32; o > 0; o >>= 1) { if (t < o) red[t] += red[t + o]; __syncthreads(); }
    float inv = 1.f / red[0]; __syncthreads();
    sc[t] = e * inv; __syncthreads();

    half_t* orow = oh + ((size_t)n * FDIM + qrow) * CP;
    for (int d = t; d < HDD; d += 64) {
        const half_t* vp = base + 2048 + h * HDD + d;
        float o = 0.f;
        for (int k = 0; k < FDIM; ++k) o += sc[k] * (float)vp[(size_t)k * 3072];
        orow[h * HDD + d] = (half_t)o;
    }
    if (h == 0 && t < (CP - CDIM)) orow[CDIM + t] = (half_t)0.f;  // zero pad cols
}

// ---------------------------------------------------------------------------
// x = LayerNorm(x + y) over the 1000 valid channels; updates x32 and xh.
// ---------------------------------------------------------------------------
__global__ __launch_bounds__(256) void k_add_ln(
    float* __restrict__ x32, half_t* __restrict__ xh,
    const float* __restrict__ y, const float* __restrict__ g,
    const float* __restrict__ be)
{
    int row = blockIdx.x, t = threadIdx.x;
    float* xr = x32 + (size_t)row * CP;
    const float* yr = y + (size_t)row * CP;
    __shared__ float red[256];

    float s = 0.f;
    for (int c = t; c < CDIM; c += 256) s += xr[c] + yr[c];
    red[t] = s; __syncthreads();
    for (int o = 128; o > 0; o >>= 1) { if (t < o) red[t] += red[t + o]; __syncthreads(); }
    float mean = red[0] * (1.f / CDIM); __syncthreads();

    float v = 0.f;
    for (int c = t; c < CDIM; c += 256) { float d = xr[c] + yr[c] - mean; v += d * d; }
    red[t] = v; __syncthreads();
    for (int o = 128; o > 0; o >>= 1) { if (t < o) red[t] += red[t + o]; __syncthreads(); }
    float rstd = rsqrtf(red[0] * (1.f / CDIM) + EPSV); __syncthreads();

    for (int c = t; c < CP; c += 256) {
        float val = 0.f;
        if (c < CDIM) val = (xr[c] + yr[c] - mean) * rstd * g[c] + be[c];
        xr[c] = val;
        xh[(size_t)row * CP + c] = (half_t)val;
    }
}

// Q[bq, d] = emb_table[concept_idx[bq]] @ w_q  (tiny: 16 x (256 -> 1024))
__global__ __launch_bounds__(256) void k_q(
    const int* __restrict__ cidx, const float* __restrict__ emb,
    const float* __restrict__ wq, float* __restrict__ Qf)
{
    int bq = blockIdx.x, t = threadIdx.x;
    int idx = cidx[bq];
    const float* er = emb + (size_t)idx * EDIM;
    for (int d = t; d < DKDIM; d += 256) {
        float s = 0.f;
        for (int e = 0; e < EDIM; ++e) s += er[e] * wq[(size_t)e * DKDIM + d];
        Qf[(size_t)bq * DKDIM + d] = s;
    }
}

// ---------------------------------------------------------------------------
// Query-focused attention: block per (b, q, s); softmax over F=64 frames
// (reference softmaxes over the last axis F only).  64 threads = frames.
// ---------------------------------------------------------------------------
__global__ __launch_bounds__(64) void k_qattn(
    const float* __restrict__ Qf, const half_t* __restrict__ Kf,
    const half_t* __restrict__ Vf, half_t* __restrict__ att)
{
    const float scale = 0.03125f;  // 1/sqrt(1024)
    int blk = blockIdx.x;
    int s = blk % SDIM; blk /= SDIM;
    int q = blk % 2;
    int b = blk / 2;
    int t = threadIdx.x;           // frame index

    size_t rbase = (size_t)(b * SDIM + s) * FDIM;
    const float* qp = Qf + (size_t)(b * 2 + q) * DKDIM;
    const half_t* kp = Kf + (rbase + t) * DKDIM;
    float dot = 0.f;
    for (int d = 0; d < DKDIM; ++d) dot += qp[d] * (float)kp[d];
    float sval = dot * scale;

    __shared__ float sc[64];
    __shared__ float red[64];
    red[t] = sval; __syncthreads();
    for (int o = 32; o > 0; o >>= 1) { if (t < o) red[t] = fmaxf(red[t], red[t + o]); __syncthreads(); }
    float mx = red[0]; __syncthreads();
    float e = expf(sval - mx);
    red[t] = e; __syncthreads();
    for (int o = 32; o > 0; o >>= 1) { if (t < o) red[t] += red[t + o]; __syncthreads(); }
    float inv = 1.f / red[0]; __syncthreads();
    sc[t] = e * inv; __syncthreads();

    half_t* orow = att + (size_t)(b * SDIM + s) * (2 * DKDIM) + q * DKDIM;
    for (int d = t; d < DKDIM; d += 64) {
        float o = 0.f;
        for (int f = 0; f < FDIM; ++f) o += sc[f] * (float)Vf[(rbase + f) * DKDIM + d];
        orow[d] = (half_t)o;
    }
}

// out[row] = sigmoid(h2[row] . mw3 + mb3)
__global__ __launch_bounds__(256) void k_head(
    const float* __restrict__ h2, const float* __restrict__ w3,
    const float* __restrict__ b3, float* __restrict__ out)
{
    int row = blockIdx.x, t = threadIdx.x;
    __shared__ float red[256];
    float s = 0.f;
    for (int d = t; d < DKDIM; d += 256) s += h2[(size_t)row * DKDIM + d] * w3[d];
    red[t] = s; __syncthreads();
    for (int o = 128; o > 0; o >>= 1) { if (t < o) red[t] += red[t + o]; __syncthreads(); }
    if (t == 0) {
        float x = red[0] + b3[0];
        out[row] = 1.f / (1.f + expf(-x));
    }
}

// ---------------------------------------------------------------------------
extern "C" void kernel_launch(void* const* d_in, const int* in_sizes, int n_in,
                              void* d_out, int out_size, void* d_ws, size_t ws_size,
                              hipStream_t stream)
{
    (void)in_sizes; (void)n_in; (void)out_size; (void)ws_size;
    const float* frame  = (const float*)d_in[0];
    const int*   cidx   = (const int*)  d_in[1];
    const float* emb    = (const float*)d_in[2];
    const float* tw_qkv = (const float*)d_in[3];
    const float* tb_qkv = (const float*)d_in[4];
    const float* tw_o   = (const float*)d_in[5];
    const float* tb_o   = (const float*)d_in[6];
    const float* tg1    = (const float*)d_in[7];
    const float* tbe1   = (const float*)d_in[8];
    const float* tw_ff1 = (const float*)d_in[9];
    const float* tb_ff1 = (const float*)d_in[10];
    const float* tw_ff2 = (const float*)d_in[11];
    const float* tb_ff2 = (const float*)d_in[12];
    const float* tg2    = (const float*)d_in[13];
    const float* tbe2   = (const float*)d_in[14];
    const float* w_q    = (const float*)d_in[15];
    const float* w_k    = (const float*)d_in[16];
    const float* w_v    = (const float*)d_in[17];
    const float* mw1    = (const float*)d_in[18];
    const float* mb1    = (const float*)d_in[19];
    const float* mw2    = (const float*)d_in[20];
    const float* mb2    = (const float*)d_in[21];
    const float* mw3    = (const float*)d_in[22];
    const float* mb3    = (const float*)d_in[23];
    float* out = (float*)d_out;

    // ---- workspace bump allocator (256B aligned) ----
    char* p = (char*)d_ws;
    auto alloc = [&](size_t bytes) -> void* {
        char* r = p; p += (bytes + 255) & ~(size_t)255; return (void*)r;
    };
    half_t* Xh   = (half_t*)alloc((size_t)NROW * CP * 2);
    float*  X32  = (float*) alloc((size_t)NROW * CP * 4);
    half_t* QKVh = (half_t*)alloc((size_t)NROW * 3072 * 2);
    half_t* Oh   = (half_t*)alloc((size_t)NROW * CP * 2);
    float*  Yf   = (float*) alloc((size_t)NROW * CP * 4);
    half_t* Hh   = (half_t*)alloc((size_t)NROW * DFFD * 2);
    half_t* Kf   = (half_t*)alloc((size_t)NROW * DKDIM * 2);
    half_t* Vf   = (half_t*)alloc((size_t)NROW * DKDIM * 2);
    float*  Qf   = (float*) alloc((size_t)16 * DKDIM * 4);
    half_t* Atth = (half_t*)alloc((size_t)256 * 2 * DKDIM * 2);
    half_t* H1h  = (half_t*)alloc((size_t)256 * DFFD * 2);
    float*  H2f  = (float*) alloc((size_t)256 * DKDIM * 4);
    half_t* Wqkv = (half_t*)alloc((size_t)CP * 3072 * 2);
    half_t* Wo   = (half_t*)alloc((size_t)CP * CP * 2);
    half_t* Wf1  = (half_t*)alloc((size_t)CP * DFFD * 2);
    half_t* Wf2  = (half_t*)alloc((size_t)DFFD * CP * 2);
    half_t* Wkh  = (half_t*)alloc((size_t)CP * DKDIM * 2);
    half_t* Wvh  = (half_t*)alloc((size_t)CP * DKDIM * 2);
    half_t* M1h  = (half_t*)alloc((size_t)(2 * DKDIM) * (2 * DKDIM) * 2);
    half_t* M2h  = (half_t*)alloc((size_t)(2 * DKDIM) * DKDIM * 2);
    float*  Bqkv = (float*) alloc(3072 * 4);
    float*  Bo   = (float*) alloc(CP * 4);
    float*  Bf2  = (float*) alloc(CP * 4);

    auto packh = [&](const float* src, int lds_, half_t* dst, int ldd,
                     int Rs, int Cs, int Rd, int Cd) {
        size_t total = (size_t)Rd * Cd;
        k_pack_h<<<dim3((unsigned)((total + 255) / 256)), 256, 0, stream>>>(
            src, lds_, dst, ldd, Rs, Cs, Rd, Cd);
    };
    auto padf = [&](const float* src, float* dst, int n, int np) {
        k_pad_f32<<<dim3((np + 255) / 256), 256, 0, stream>>>(src, dst, n, np);
    };

    // input transpose + pad + f16 copy
    {
        size_t total = (size_t)NROW * CP;
        k_prep<<<dim3((unsigned)((total + 255) / 256)), 256, 0, stream>>>(frame, X32, Xh);
    }

    // ---- 6-layer transformer encoder ----
    for (int l = 0; l < LNUM; ++l) {
        for (int sec = 0; sec < 3; ++sec) {
            packh(tw_qkv + ((size_t)l * CDIM) * 3 * CDIM + sec * CDIM, 3 * CDIM,
                  Wqkv + sec * CP, 3 * CP, CDIM, CDIM, CP, CP);
            padf(tb_qkv + (size_t)l * 3 * CDIM + sec * CDIM, Bqkv + sec * CP, CDIM, CP);
        }
        packh(tw_o + (size_t)l * CDIM * CDIM, CDIM, Wo, CP, CDIM, CDIM, CP, CP);
        padf(tb_o + (size_t)l * CDIM, Bo, CDIM, CP);
        packh(tw_ff1 + (size_t)l * CDIM * DFFD, DFFD, Wf1, DFFD, CDIM, DFFD, CP, DFFD);
        packh(tw_ff2 + (size_t)l * DFFD * CDIM, CDIM, Wf2, CP, DFFD, CDIM, DFFD, CP);
        padf(tb_ff2 + (size_t)l * CDIM, Bf2, CDIM, CP);

        // QKV projection: [16384,1024] x [1024,3072]
        k_gemm<<<dim3(3 * CP / 16, NROW / 256), 256, 0, stream>>>(
            Xh, Wqkv, Bqkv, nullptr, QKVh, NROW, 3 * CP, CP, 0);
        // self-attention (softmax over 64 keys per (n,h,q))
        k_attn<<<dim3(BDIM * SDIM * HDIM * FDIM), 64, 0, stream>>>(QKVh, Oh);
        // output projection
        k_gemm<<<dim3(CP / 16, NROW / 256), 256, 0, stream>>>(
            Oh, Wo, Bo, Yf, nullptr, NROW, CP, CP, 0);
        k_add_ln<<<dim3(NROW), 256, 0, stream>>>(
            X32, Xh, Yf, tg1 + (size_t)l * CDIM, tbe1 + (size_t)l * CDIM);
        // FFN
        k_gemm<<<dim3(DFFD / 16, NROW / 256), 256, 0, stream>>>(
            Xh, Wf1, tb_ff1 + (size_t)l * DFFD, nullptr, Hh, NROW, DFFD, CP, 1);
        k_gemm<<<dim3(CP / 16, NROW / 256), 256, 0, stream>>>(
            Hh, Wf2, Bf2, Yf, nullptr, NROW, CP, DFFD, 0);
        k_add_ln<<<dim3(NROW), 256, 0, stream>>>(
            X32, Xh, Yf, tg2 + (size_t)l * CDIM, tbe2 + (size_t)l * CDIM);
    }

    // ---- query-focused attention + MLP head ----
    packh(w_k, DKDIM, Wkh, DKDIM, CDIM, DKDIM, CP, DKDIM);
    packh(w_v, DKDIM, Wvh, DKDIM, CDIM, DKDIM, CP, DKDIM);
    packh(mw1, 2 * DKDIM, M1h, 2 * DKDIM, 2 * DKDIM, 2 * DKDIM, 2 * DKDIM, 2 * DKDIM);
    packh(mw2, DKDIM, M2h, DKDIM, 2 * DKDIM, DKDIM, 2 * DKDIM, DKDIM);

    k_gemm<<<dim3(DKDIM / 16, NROW / 256), 256, 0, stream>>>(
        Xh, Wkh, nullptr, nullptr, Kf, NROW, DKDIM, CP, 0);
    k_gemm<<<dim3(DKDIM / 16, NROW / 256), 256, 0, stream>>>(
        Xh, Wvh, nullptr, nullptr, Vf, NROW, DKDIM, CP, 0);
    k_q<<<dim3(BDIM * 2), 256, 0, stream>>>(cidx, emb, w_q, Qf);
    k_qattn<<<dim3(BDIM * 2 * SDIM), 64, 0, stream>>>(Qf, Kf, Vf, Atth);

    k_gemm<<<dim3((2 * DKDIM) / 16, 1), 256, 0, stream>>>(
        Atth, M1h, mb1, nullptr, H1h, 256, 2 * DKDIM, 2 * DKDIM, 1);
    k_gemm<<<dim3(DKDIM / 16, 1), 256, 0, stream>>>(
        H1h, M2h, mb2, H2f, nullptr, 256, DKDIM, 2 * DKDIM, 1);
    k_head<<<dim3(256), 256, 0, stream>>>(H2f, mw3, mb3, out);
}
